// RobustSAMCell_17755394801939
// MI455X (gfx1250) — compile-verified, matrix operand order checked by reading, not run on hardware
//
#include <hip/hip_runtime.h>
#include <hip/hip_bf16.h>

typedef __attribute__((ext_vector_type(16))) _Float16 v16h;
typedef __attribute__((ext_vector_type(8)))  _Float16 v8h;
typedef __attribute__((ext_vector_type(4)))  _Float16 v4h;
typedef __attribute__((ext_vector_type(8)))  float    v8f;
typedef __attribute__((ext_vector_type(4)))  int      v4i;

#define DN   512
#define EN   256
#define TN   128
#define KIN  896   // D+E+T
#define KP   5
#define BM   32
#define BSTRIDE 40  // halves per LDS row (32 data + 8 pad) = 80B, 16B aligned, conflict-free

// gfx1250 async global->LDS copy (ASYNCcnt-tracked), if this toolchain exposes it.
#if defined(__has_builtin)
#  if __has_builtin(__builtin_amdgcn_global_load_async_to_lds_b128)
#    define HAS_ASYNC_LDS 1
#  endif
#endif
#ifndef HAS_ASYNC_LDS
#  define HAS_ASYNC_LDS 0
#endif

#if HAS_ASYNC_LDS
typedef __attribute__((address_space(1))) v4i as1_v4i;  // global int4
typedef __attribute__((address_space(3))) v4i as3_v4i;  // LDS int4
#endif

__device__ __forceinline__ float clampf(float x, float lo, float hi) {
  return fminf(fmaxf(x, lo), hi);
}

__device__ __forceinline__ float wsum(float v) {
  v += __shfl_xor(v, 16, 32);
  v += __shfl_xor(v, 8, 32);
  v += __shfl_xor(v, 4, 32);
  v += __shfl_xor(v, 2, 32);
  v += __shfl_xor(v, 1, 32);
  return v;
}

// Kernel 0: convert Wq fp32 [512][896] -> f16 same layout (this is B^T: B[k][d]=Wq[d][k])
__global__ void wq_to_f16(const float* __restrict__ Wq, _Float16* __restrict__ Wqh, int n) {
  int i = blockIdx.x * blockDim.x + threadIdx.x;
  if (i < n) Wqh[i] = (_Float16)Wq[i];
}

// Fused: q-projection GEMM (WMMA f16->f32) + LN/tanh + prototype attention + gate + final LN.
// One block = 32 node rows x all 512 output cols. 8 waves (wave32) in 2x4 grid,
// each wave owns 16 rows x 128 cols = 8 C tiles of 16x16.
__global__ __launch_bounds__(256)
void robust_sam_fused(const float* __restrict__ raw,
                      const float* __restrict__ edge,
                      const float* __restrict__ timeE,
                      const float* __restrict__ proto,
                      const _Float16* __restrict__ Wqh,
                      const float* __restrict__ bq,
                      const float* __restrict__ Wg,
                      const float* __restrict__ bg,
                      const float* __restrict__ gamma,
                      const float* __restrict__ beta,
                      const float* __restrict__ temp,
                      float* __restrict__ out) {
  __shared__ char smem[65536];
  _Float16* Bst = (_Float16*)smem;            // [512][BSTRIDE] halves  (40960 B)
  _Float16* Ast = (_Float16*)(smem + 40960);  // [32][BSTRIDE]  halves  ( 2560 B)
  float*    zbuf = (float*)smem;              // [32][512] fp32 (aliases staging after K loop)

  const int tid  = threadIdx.x;
  const int lane = tid & 31;
  const int wave = tid >> 5;
  const int wr   = wave >> 2;          // 0..1 : row group of 16
  const int wc   = wave & 3;           // 0..3 : col group of 128
  const int m16  = lane & 15;
  const int cb   = (lane < 16) ? 0 : 8; // per-ISA half-wave K base
  const size_t row0 = (size_t)blockIdx.x * BM;

  v8f acc[8] = {};

  for (int kc = 0; kc < KIN; kc += 32) {
    __syncthreads();
    // ---- stage B chunk: Wqh[d][kc..kc+31] -> Bst[d][0..31]; 2 rows per thread, 4x16B each
    {
      int d = tid * 2;
      #pragma unroll
      for (int rr = 0; rr < 2; ++rr, ++d) {
        const _Float16* src = Wqh + (size_t)d * KIN + kc;
        _Float16* dst = Bst + d * BSTRIDE;
#if HAS_ASYNC_LDS
        #pragma unroll
        for (int c = 0; c < 4; ++c) {
          __builtin_amdgcn_global_load_async_to_lds_b128(
              (as1_v4i*)(src + c * 8), (as3_v4i*)(dst + c * 8), 0, 0);
        }
#else
        const uint4* s4 = (const uint4*)src;
        uint4* d4 = (uint4*)dst;
        d4[0] = s4[0]; d4[1] = s4[1]; d4[2] = s4[2]; d4[3] = s4[3];
#endif
      }
    }
    // ---- stage A chunk with clip(+-50) + cvt to f16 (concat boundaries align to 32)
    {
      int r  = tid >> 3;          // 0..31
      int k4 = (tid & 7) * 4;     // 0..28
      const float* src; int stride, off;
      if (kc < DN)            { src = raw;   stride = DN; off = kc; }
      else if (kc < DN + EN)  { src = edge;  stride = EN; off = kc - DN; }
      else                    { src = timeE; stride = TN; off = kc - DN - EN; }
      float4 v = *(const float4*)(src + (row0 + r) * (size_t)stride + off + k4);
      v4h h;
      h[0] = (_Float16)clampf(v.x, -50.f, 50.f);
      h[1] = (_Float16)clampf(v.y, -50.f, 50.f);
      h[2] = (_Float16)clampf(v.z, -50.f, 50.f);
      h[3] = (_Float16)clampf(v.w, -50.f, 50.f);
      *(v4h*)(Ast + r * BSTRIDE + k4) = h;
    }
#if HAS_ASYNC_LDS
    asm volatile("s_wait_asynccnt 0x0" ::: "memory");  // our async LDS writes landed
#endif
    __syncthreads();
    // ---- fragments + 8 WMMAs per wave
    v8h alo = *(const v8h*)(Ast + (wr * 16 + m16) * BSTRIDE + cb);
    v8h ahi = *(const v8h*)(Ast + (wr * 16 + m16) * BSTRIDE + cb + 16);
    v16h a = __builtin_shufflevector(alo, ahi, 0,1,2,3,4,5,6,7,8,9,10,11,12,13,14,15);
    #pragma unroll
    for (int t = 0; t < 8; ++t) {
      int n = wc * 128 + t * 16 + m16;
      v8h blo = *(const v8h*)(Bst + n * BSTRIDE + cb);
      v8h bhi = *(const v8h*)(Bst + n * BSTRIDE + cb + 16);
      v16h b = __builtin_shufflevector(blo, bhi, 0,1,2,3,4,5,6,7,8,9,10,11,12,13,14,15);
      acc[t] = __builtin_amdgcn_wmma_f32_16x16x32_f16(false, a, false, b,
                                                      (short)0, acc[t], false, false);
    }
  }

  __syncthreads();  // staging reads done -> safe to overwrite with z
  // ---- scatter accumulators to z-buffer (C layout: VGPR j -> M=j (+8 for upper half-wave))
  #pragma unroll
  for (int t = 0; t < 8; ++t) {
    #pragma unroll
    for (int j = 0; j < 8; ++j) {
      int row = wr * 16 + ((lane < 16) ? 0 : 8) + j;
      int col = wc * 128 + t * 16 + m16;
      zbuf[row * DN + col] = acc[t][j];
    }
  }
  __syncthreads();

  // ================= epilogue: one wave per node row, 4 rows per wave =================
  const float tinv = 1.0f / (clampf(temp[0], 0.5f, 5.0f) + 1e-4f);
  const float bgv  = bg[0];
  for (int rr = 0; rr < 4; ++rr) {
    const int row = wave * 4 + rr;
    const size_t grow = row0 + row;
    __builtin_prefetch(proto + grow * (size_t)(KP * DN), 0, 0);

    // ---- LN1 + tanh -> query
    float z[16];
    #pragma unroll
    for (int i = 0; i < 16; ++i) {
      int c = i * 32 + lane;
      z[i] = zbuf[row * DN + c] + bq[c];
    }
    float s = 0.f;
    #pragma unroll
    for (int i = 0; i < 16; ++i) s += z[i];
    float mu = wsum(s) * (1.0f / DN);
    float ss = 0.f;
    #pragma unroll
    for (int i = 0; i < 16; ++i) { float d = z[i] - mu; ss += d * d; }
    float rstd = rsqrtf(wsum(ss) * (1.0f / DN) + 1e-6f);
    float qss = 0.f;
    #pragma unroll
    for (int i = 0; i < 16; ++i) {
      int c = i * 32 + lane;
      float q = tanhf((z[i] - mu) * rstd * gamma[c] + beta[c]); // |q|<1 -> +-20 clip no-op
      z[i] = q;
      qss += q * q;
    }
    float qinv = 1.0f / fmaxf(sqrtf(wsum(qss)), 1e-6f);

    // ---- cosine sims vs 5 prototypes (kept in regs: single HBM read of prototypes)
    float p[KP][16], sim[KP];
    #pragma unroll
    for (int k = 0; k < KP; ++k) {
      const float* pp = proto + (grow * KP + k) * (size_t)DN;
      float dt = 0.f, pss = 0.f;
      #pragma unroll
      for (int i = 0; i < 16; ++i) {
        float v = pp[i * 32 + lane];
        p[k][i] = v;
        float vc = clampf(v, -20.f, 20.f);
        pss += vc * vc;
        dt  += z[i] * vc;
      }
      pss = wsum(pss);
      dt  = wsum(dt);
      float pinv = 1.0f / fmaxf(sqrtf(pss), 1e-6f);
      sim[k] = clampf(dt * qinv * pinv, -15.f, 15.f) * tinv;
    }
    // ---- softmax over K=5 (replicated per lane; all lanes hold identical sims)
    float mx = sim[0];
    #pragma unroll
    for (int k = 1; k < KP; ++k) mx = fmaxf(mx, sim[k]);
    float e[KP], se = 0.f;
    #pragma unroll
    for (int k = 0; k < KP; ++k) { e[k] = expf(sim[k] - mx); se += e[k]; }
    float attn[KP];
    #pragma unroll
    for (int k = 0; k < KP; ++k) attn[k] = clampf(e[k] / se, 0.f, 1.f);

    // ---- candidate + raw reload + gate logit
    float rawv[16], cand[16];
    float gl = 0.f;
    #pragma unroll
    for (int i = 0; i < 16; ++i) {
      int c = i * 32 + lane;
      float cs = 0.f;
      #pragma unroll
      for (int k = 0; k < KP; ++k) cs += attn[k] * p[k][i];
      cand[i] = clampf(cs, -5.f, 5.f);
      rawv[i] = clampf(raw[grow * DN + c], -50.f, 50.f);
      gl += clampf(rawv[i], -30.f, 30.f) * Wg[c]
          + cand[i] * Wg[DN + c];                  // |cand|<=5 -> +-30 clip no-op
    }
    #pragma unroll
    for (int i = 0; i < 4; ++i) {
      int c = i * 32 + lane;
      gl += clampf(timeE[grow * TN + c], -30.f, 30.f) * Wg[2 * DN + c];
    }
    gl = clampf(wsum(gl) + bgv, -10.f, 10.f);
    float g = 1.0f / (1.0f + expf(-gl));

    // ---- residual mix + final LN + clamp
    float us = 0.f;
    #pragma unroll
    for (int i = 0; i < 16; ++i) {
      float gated = (1.0f - g) * rawv[i] + g * cand[i];
      float u = 0.8f * rawv[i] + 0.2f * gated;
      rawv[i] = u;
      us += u;
    }
    float mu2 = wsum(us) * (1.0f / DN);
    float ss2 = 0.f;
    #pragma unroll
    for (int i = 0; i < 16; ++i) { float d = rawv[i] - mu2; ss2 += d * d; }
    float rstd2 = rsqrtf(wsum(ss2) * (1.0f / DN) + 1e-6f);
    #pragma unroll
    for (int i = 0; i < 16; ++i) {
      int c = i * 32 + lane;
      out[grow * DN + c] = clampf((rawv[i] - mu2) * rstd2 * gamma[c] + beta[c], -10.f, 10.f);
    }
  }
}

extern "C" void kernel_launch(void* const* d_in, const int* in_sizes, int n_in,
                              void* d_out, int out_size, void* d_ws, size_t ws_size,
                              hipStream_t stream) {
  const float* raw   = (const float*)d_in[0];
  // d_in[1] = node_features: unused by the reference
  const float* edge  = (const float*)d_in[2];
  const float* timeE = (const float*)d_in[3];
  const float* proto = (const float*)d_in[4];
  const float* Wq    = (const float*)d_in[5];
  const float* bq    = (const float*)d_in[6];
  const float* Wg    = (const float*)d_in[7];
  const float* bg    = (const float*)d_in[8];
  const float* gamma = (const float*)d_in[9];
  const float* beta  = (const float*)d_in[10];
  const float* temp  = (const float*)d_in[11];
  float* out = (float*)d_out;

  _Float16* Wqh = (_Float16*)d_ws;            // 512*896*2 = 0.92 MB scratch
  const int nWq = DN * KIN;
  wq_to_f16<<<(nWq + 255) / 256, 256, 0, stream>>>(Wq, Wqh, nWq);

  const int N = 65536;
  robust_sam_fused<<<N / BM, 256, 0, stream>>>(raw, edge, timeE, proto, Wqh, bq,
                                               Wg, bg, gamma, beta, temp, out);
}
